// TennisGraphNN_55087250538978
// MI455X (gfx1250) — matrix operand order repaired and checked by reading, not compiled, not run on hardware
//
#include <hip/hip_runtime.h>
#include <hip/hip_bf16.h>

#define DIM 256
#define BATCH 50000
#define N_NODES (2 * BATCH)
#define N_EDGES 320000

typedef __attribute__((ext_vector_type(16))) __bf16 v16bf;
typedef __attribute__((ext_vector_type(8)))  float  v8f;

static __device__ __forceinline__ __bf16 to_bf16(float f) {
    // round-to-nearest-even f32 -> bf16
    unsigned u = __float_as_uint(f);
    unsigned r = (u + 0x7FFFu + ((u >> 16) & 1u)) >> 16;
    unsigned short s = (unsigned short)r;
    __bf16 h;
    __builtin_memcpy(&h, &s, 2);
    return h;
}

// ---------------- weight pre-pack into WMMA B-fragment order ----------------
//   Bpk[ ((ct*(K/32)+kt)*32 + lane)*16 + e ] = bf16( W[ kk*N + n ] )
//   n = ct*16 + (lane&15);  kk = kt*32 + 16*(lane>>4) + e   (contiguous k order, same as A stage)
__global__ void pack_w_kernel(const float* __restrict__ W, __bf16* __restrict__ Bpk,
                              int K, int N) {
    unsigned idx = blockIdx.x * blockDim.x + threadIdx.x;   // exactly K*N threads
    int e      = (int)(idx & 15);
    int lane   = (int)((idx >> 4) & 31);
    int blk    = (int)(idx >> 9);            // ct*(K/32) + kt
    int ksteps = K >> 5;
    int ct     = blk / ksteps;
    int kt     = blk - ct * ksteps;
    int n      = ct * 16 + (lane & 15);
    int kk     = kt * 32 + ((lane >> 4) << 4) + e;
    Bpk[idx] = to_bf16(W[(unsigned)kk * N + n]);
}

// ---------------- small utility kernels ----------------

__global__ void gather_emb_kernel(const float* __restrict__ emb, const int* __restrict__ p1,
                                  const int* __restrict__ p2, float* __restrict__ x) {
    unsigned idx = blockIdx.x * blockDim.x + threadIdx.x; // N_NODES*DIM threads exactly
    int i = (int)(idx >> 8);
    int d = (int)(idx & (DIM - 1));
    int row = (i < BATCH) ? p1[i] : p2[i - BATCH];
    x[idx] = emb[(unsigned)row * DIM + d];
}

__global__ void init_deg_kernel(float* __restrict__ deg) {
    int i = blockIdx.x * blockDim.x + threadIdx.x;
    if (i < N_NODES) deg[i] = 1.0f; // self-loop contribution
}

__global__ void add_deg_kernel(float* __restrict__ deg, const int* __restrict__ dst) {
    int e = blockIdx.x * blockDim.x + threadIdx.x;
    if (e < N_EDGES) atomicAdd(&deg[dst[e]], 1.0f);
}

__global__ void finalize_dinv_kernel(float* __restrict__ deg) {
    int i = blockIdx.x * blockDim.x + threadIdx.x;
    if (i < N_NODES) {
        float d = deg[i];
        deg[i] = (d > 0.0f) ? rsqrtf(d) : 0.0f;
    }
}

__global__ void zero_kernel(float* __restrict__ p) {
    unsigned i = blockIdx.x * blockDim.x + threadIdx.x;
    p[i] = 0.0f;
}

// Edge scatter: out[dst] += h[src] * dinv[src]*dinv[dst].
// Virtual edges [N_EDGES, N_EDGES+N_NODES) are the self-loops.
__global__ void scatter_kernel(const float* __restrict__ h, float* __restrict__ out,
                               const int* __restrict__ src, const int* __restrict__ dst,
                               const float* __restrict__ dinv) {
    unsigned t = blockIdx.x * blockDim.x + threadIdx.x; // (N_EDGES+N_NODES)*DIM threads
    int e = (int)(t >> 8);
    int d = (int)(t & (DIM - 1));
    int s, dd;
    if (e < N_EDGES) { s = src[e]; dd = dst[e]; }
    else             { s = dd = e - N_EDGES; }
    float nrm = dinv[s] * dinv[dd];
    atomicAdd(&out[(unsigned)dd * DIM + d], h[(unsigned)s * DIM + d] * nrm);
}

// ---------------- WMMA GEMM with fused A-side bias/ReLU ----------------
// C = act( prepA(A) @ W + bias ), W pre-packed via pack_w_kernel.
// MODE==1 : A is the virtual concat( x[:BATCH], x[BATCH:], axis=1 ); storage N_NODES x 256.
// APRE    : 0 none, 1 A += abias[feat], 2 A = relu(A + abias[feat])   (feat = kk & 255)
// ACT/HAS_BIAS : epilogue ReLU / column bias.
// NTILES  : 16-col output tiles per wave; with 8 waves a block covers 128*NTILES cols,
//           chosen so one block spans all N columns -> A streamed exactly once.
// All waves share one 16-row A tile staged through double-buffered LDS in fragment order.
template<int MODE, int APRE, int ACT, int HAS_BIAS, int NTILES>
__global__ __launch_bounds__(256) void gemm_wmma_kernel(
    const float* __restrict__ A, const __bf16* __restrict__ Bpk, float* __restrict__ C,
    const float* __restrict__ bias, const float* __restrict__ abias, int K, int N)
{
    __shared__ __align__(32) __bf16 lA[2][16 * 32]; // double-buffered A tile, fragment order

    const int tid  = threadIdx.x;
    const int wave = tid >> 5;
    const int lane = tid & 31;
    const int m16  = lane & 15;
    const int hi   = lane >> 4;

    const int rowBase  = blockIdx.x * 16;
    const int colTile0 = (blockIdx.y * 8 + wave) * NTILES;  // first 16-col tile of this wave
    const int ksteps   = K >> 5;

    // A-stage: this thread supplies rows r and r+8 at k-offset c of each step
    const int r = tid >> 5;                      // 0..7
    const int c = tid & 31;                      // 0..31
    // LDS halfword index for (row r, k-offset c):
    //   lane = r + 16*(c>>4), elem = c&15
    const int fidx = (r + (c >> 4) * 16) * 16 + (c & 15);
    const unsigned aBase = (unsigned)(rowBase + r) * (MODE == 1 ? DIM : K) + (MODE == 1 ? 0 : c);

    v8f acc[NTILES];
#pragma unroll
    for (int t = 0; t < NTILES; ++t) acc[t] = (v8f){};

    const unsigned tileStride = (unsigned)ksteps * 512;     // packed elems per col tile
    unsigned bIdx = (unsigned)colTile0 * tileStride + (unsigned)lane * 16;

    for (int kt = 0; kt < ksteps; ++kt) {
        // ---- cooperative A stage: coalesced f32 loads, bf16 convert, pre-swizzle ----
        float v0, v1;
        if (MODE == 1) {
            int kk = kt * 32 + c;
            unsigned off = (kk < DIM) ? (aBase + kk)
                                      : (aBase + (unsigned)BATCH * DIM + (kk - DIM));
            v0 = A[off];
            v1 = A[off + 8u * DIM];
        } else {
            unsigned off = aBase + (unsigned)kt * 32;
            v0 = A[off];
            v1 = A[off + 8u * K];
        }
        if (APRE) {
            float ab = abias[(kt * 32 + c) & (DIM - 1)];
            v0 += ab; v1 += ab;
            if (APRE == 2) { v0 = fmaxf(v0, 0.0f); v1 = fmaxf(v1, 0.0f); }
        }
        __bf16* dstL = lA[kt & 1];
        dstL[fidx]       = to_bf16(v0);
        dstL[fidx + 128] = to_bf16(v1);      // row r+8 -> +8*16 halfwords
        __syncthreads();

        v16bf a = *(const v16bf*)&lA[kt & 1][lane * 16];
#pragma unroll
        for (int t = 0; t < NTILES; ++t) {
            v16bf b = *(const v16bf*)&Bpk[bIdx + (unsigned)t * tileStride];
            acc[t] = __builtin_amdgcn_wmma_f32_16x16x32_bf16(
                /*neg_a=*/false, a, /*neg_b=*/false, b,
                /*c_mod=*/(short)0, acc[t], /*reuse_a=*/false, /*reuse_b=*/false);
        }
        __builtin_prefetch(&Bpk[bIdx + 512], 0, 0);  // next k-step of tile 0 (buffers padded)
        bIdx += 512;
        // no second barrier: 2-deep LDS rotation makes the next write race-free
    }

    // ---- epilogue: C/D 16x16 f32 layout: element rr -> M = 8*hi + rr, N = m16 ----
#pragma unroll
    for (int t = 0; t < NTILES; ++t) {
        const int nOut = (colTile0 + t) * 16 + m16;
        float bv = HAS_BIAS ? bias[nOut] : 0.0f;
#pragma unroll
        for (int rr = 0; rr < 8; ++rr) {
            const int mOut = rowBase + hi * 8 + rr;
            float v = acc[t][rr];
            if (HAS_BIAS) v += bv;
            if (ACT) v = fmaxf(v, 0.0f);
            C[(unsigned)mOut * N + nOut] = v;
        }
    }
}

// Final 256->1 dot per row + sigmoid. One wave per row, 8 rows per block.
__global__ __launch_bounds__(256) void final_dot_kernel(
    const float* __restrict__ h2, const float* __restrict__ w,
    const float* __restrict__ b, float* __restrict__ out)
{
    const int row  = blockIdx.x * 8 + (threadIdx.x >> 5);
    const int lane = threadIdx.x & 31;
    float s = 0.0f;
#pragma unroll
    for (int d = lane; d < DIM; d += 32)
        s += h2[(unsigned)row * DIM + d] * w[d];
#pragma unroll
    for (int off = 16; off > 0; off >>= 1)
        s += __shfl_xor(s, off, 32);
    if (lane == 0)
        out[row] = 1.0f / (1.0f + __expf(-(s + b[0])));
}

// ---------------- host launcher ----------------

extern "C" void kernel_launch(void* const* d_in, const int* in_sizes, int n_in,
                              void* d_out, int out_size, void* d_ws, size_t ws_size,
                              hipStream_t stream) {
    (void)in_sizes; (void)n_in; (void)out_size; (void)ws_size;

    const int*   p1   = (const int*)d_in[0];
    const int*   p2   = (const int*)d_in[1];
    const int*   esrc = (const int*)d_in[2];
    const int*   edst = esrc + N_EDGES;
    const float* emb  = (const float*)d_in[3];
    const float* W1  = (const float*)d_in[4];   const float* b1  = (const float*)d_in[5];
    const float* W2  = (const float*)d_in[6];   const float* b2  = (const float*)d_in[7];
    const float* W3  = (const float*)d_in[8];   const float* b3  = (const float*)d_in[9];
    const float* Wh1 = (const float*)d_in[10];  const float* bh1 = (const float*)d_in[11];
    const float* Wh2 = (const float*)d_in[12];  const float* bh2 = (const float*)d_in[13];
    const float* Wh3 = (const float*)d_in[14];  const float* bh3 = (const float*)d_in[15];
    float* out = (float*)d_out;

    // workspace layout (packed weights padded by 512 bf16 for unconditional prefetch)
    const int PAD = 512;
    float*  bufA = (float*)d_ws;                       // N_NODES*DIM f32
    float*  bufB = bufA + (long)N_NODES * DIM;         // N_NODES*DIM f32
    float*  dinv = bufB + (long)N_NODES * DIM;         // N_NODES f32
    __bf16* W1p  = (__bf16*)(dinv + N_NODES);
    __bf16* W2p  = W1p + DIM * DIM + PAD;
    __bf16* W3p  = W2p + DIM * DIM + PAD;
    __bf16* Wh1p = W3p + DIM * DIM + PAD;              // 512*512
    __bf16* Wh2p = Wh1p + 2 * DIM * 512 + PAD;         // 512*256

    // 1) pre-pack weights into WMMA B-fragment order (tiny, resident in L2)
    pack_w_kernel<<<(DIM * DIM) / 256, 256, 0, stream>>>(W1, W1p, DIM, DIM);
    pack_w_kernel<<<(DIM * DIM) / 256, 256, 0, stream>>>(W2, W2p, DIM, DIM);
    pack_w_kernel<<<(DIM * DIM) / 256, 256, 0, stream>>>(W3, W3p, DIM, DIM);
    pack_w_kernel<<<(2 * DIM * 512) / 256, 256, 0, stream>>>(Wh1, Wh1p, 512, 512);
    pack_w_kernel<<<(512 * DIM) / 256, 256, 0, stream>>>(Wh2, Wh2p, 512, DIM);

    // 2) node features = gathered embeddings
    gather_emb_kernel<<<N_NODES * DIM / 256, 256, 0, stream>>>(emb, p1, p2, bufA);

    // 3) degrees -> dinv
    init_deg_kernel<<<(N_NODES + 255) / 256, 256, 0, stream>>>(dinv);
    add_deg_kernel<<<(N_EDGES + 255) / 256, 256, 0, stream>>>(dinv, edst);
    finalize_dinv_kernel<<<(N_NODES + 255) / 256, 256, 0, stream>>>(dinv);

    // 4) three GCN layers; bias+ReLU of each layer fused into the NEXT GEMM's A-stage
    const dim3 ggrid(N_NODES / 16, 1);   // NTILES=2 -> one block spans all 256 cols
    // layer 1: h = emb_x @ W1
    gemm_wmma_kernel<0, 0, 0, 0, 2><<<ggrid, 256, 0, stream>>>(
        bufA, W1p, bufB, nullptr, nullptr, DIM, DIM);
    zero_kernel<<<N_NODES * DIM / 256, 256, 0, stream>>>(bufA);
    scatter_kernel<<<(N_EDGES + N_NODES), 256, 0, stream>>>(bufB, bufA, esrc, edst, dinv);
    // layer 2: h = relu(agg1 + b1) @ W2
    gemm_wmma_kernel<0, 2, 0, 0, 2><<<ggrid, 256, 0, stream>>>(
        bufA, W2p, bufB, nullptr, b1, DIM, DIM);
    zero_kernel<<<N_NODES * DIM / 256, 256, 0, stream>>>(bufA);
    scatter_kernel<<<(N_EDGES + N_NODES), 256, 0, stream>>>(bufB, bufA, esrc, edst, dinv);
    // layer 3: h = relu(agg2 + b2) @ W3
    gemm_wmma_kernel<0, 2, 0, 0, 2><<<ggrid, 256, 0, stream>>>(
        bufA, W3p, bufB, nullptr, b2, DIM, DIM);
    zero_kernel<<<N_NODES * DIM / 256, 256, 0, stream>>>(bufA);
    scatter_kernel<<<(N_EDGES + N_NODES), 256, 0, stream>>>(bufB, bufA, esrc, edst, dinv);
    // bufA now holds raw agg3; x3 = agg3 + b3 (no relu), applied in MLP1 A-stage

    // 5) MLP head
    // h1 = relu(concat(x3[:B], x3[B:]) @ Wh1 + bh1) -> bufB (50000 x 512); NTILES=4 spans 512 cols
    gemm_wmma_kernel<1, 1, 1, 1, 4><<<dim3(BATCH / 16, 1), 256, 0, stream>>>(
        bufA, Wh1p, bufB, bh1, b3, 512, 512);
    // h2 = relu(h1 @ Wh2 + bh2) -> bufA (50000 x 256); NTILES=2 spans 256 cols
    gemm_wmma_kernel<0, 0, 1, 1, 2><<<dim3(BATCH / 16, 1), 256, 0, stream>>>(
        bufB, Wh2p, bufA, bh2, nullptr, 512, DIM);
    // out = sigmoid(h2 @ Wh3 + bh3)
    final_dot_kernel<<<BATCH / 8, 256, 0, stream>>>(bufA, Wh3, bh3, out);
}
